// HawkesPointProcess_37125697307224
// MI455X (gfx1250) — compile-verified, matrix operand order checked by reading, not run on hardware
//
#include <hip/hip_runtime.h>

// CDNA5 / gfx1250: wave32, WMMA 16x16x32 f16->f32.
typedef __attribute__((ext_vector_type(16))) _Float16 v16h;
typedef __attribute__((ext_vector_type(8)))  float    v8f;

#define MAXN    4096
#define MAXBLK  (MAXN / 16)
#define NTHREADS 256
#define NWAVES   (NTHREADS / 32)

__device__ __forceinline__ float softplus_f(float x) {
    if (x > 20.0f) return x;
    return log1pf(__expf(x));
}

__device__ __forceinline__ float wave_sum32(float v) {
    #pragma unroll
    for (int off = 16; off > 0; off >>= 1)
        v += __shfl_down(v, off, 32);
    return v;  // valid in lane 0
}

// One workgroup (8 wave32) per batch row.
//  Phase 1: waves process 16-event tiles in parallel. Per tile, the strictly
//           lower-triangular exp matrix is row-summed with one
//           V_WMMA_F32_16X16X32_F16 (B = ones). Lanes 0/16 (column N=0) hold
//           full row-sum vectors -> two ds_store_b128 park W_i in LDS, and the
//           per-tile linear-recurrence coefficients (decay_p, L_p) are stored.
//  Phase 2: Kogge-Stone scan of H_p = decay_p*H_{p-1} + L_p over 256 tiles
//           (associative pair composition, log2(256)=8 barrier steps).
//  Phase 3: 256 threads compute per-event log(lambda) in parallel; block
//           reduction via ds_add_f32 (shared atomicAdd).
__global__ __launch_bounds__(NTHREADS)
void hawkes_loglik_kernel(const float* __restrict__ event_times,
                          const float* __restrict__ input_mask,
                          const float* __restrict__ t0a,
                          const float* __restrict__ t1a,
                          const float* __restrict__ mu_raw,
                          const float* __restrict__ alpha_raw,
                          const float* __restrict__ beta_raw,
                          float* __restrict__ out,
                          int N)
{
    __shared__ __align__(16) float sW[MAXN];   // within-tile excitation W_i
    __shared__ float sAa[MAXBLK];              // decay_p
    __shared__ float sAb[MAXBLK];              // L_p, becomes H_p after scan
    __shared__ float sLog, sMsk;

    const int b    = blockIdx.x;
    const int tid  = threadIdx.x;
    const int wave = tid >> 5;
    const int lane = tid & 31;

    const float mu    = softplus_f(mu_raw[0]);
    const float alpha = softplus_f(alpha_raw[0]);
    const float beta  = softplus_f(beta_raw[0]);
    const float ab    = alpha * beta;

    const float* __restrict__ t  = event_times + (size_t)b * N;
    const float* __restrict__ mk = input_mask  + (size_t)b * N;

    const int nblk = N >> 4;

    if (tid == 0) { sLog = 0.0f; sMsk = 0.0f; }

    // A-matrix lane mapping (16-bit A 16x32, ISA 7.12.2):
    //   lanes 0-15 : M = lane,    halves 0..7 -> K = 0..7
    //   lanes 16-31: M = lane-16, halves 0..7 -> K = 8..15
    //   halves 8..15 (K = 16..31) zeroed.
    const int m     = lane & 15;
    const int kbase = (lane < 16) ? 0 : 8;

    v16h ones;
    #pragma unroll
    for (int i = 0; i < 16; ++i) ones[i] = (_Float16)1.0f;

    // ---------------- Phase 1: parallel WMMA tiles ----------------
    for (int p = wave; p < nblk; p += NWAVES) {
        const int r0 = p << 4;

        if (p + NWAVES < nblk)
            __builtin_prefetch(t + ((p + NWAVES) << 4), 0, 1);  // global_prefetch

        const float  tm = t[r0 + m];                             // row time
        const float4 c0 = *reinterpret_cast<const float4*>(t + r0 + kbase);
        const float4 c1 = *reinterpret_cast<const float4*>(t + r0 + kbase + 4);
        const float  col[8] = {c0.x, c0.y, c0.z, c0.w, c1.x, c1.y, c1.z, c1.w};

        v16h a;
        #pragma unroll
        for (int h = 0; h < 8; ++h) {
            const int k = kbase + h;
            const float e = (k < m) ? __expf(-beta * (tm - col[h])) : 0.0f;
            a[h]     = (_Float16)e;
            a[h + 8] = (_Float16)0.0f;
        }

        // Row sums: D = A x ones + 0  (uniform control flow, EXEC all-1s)
        v8f c = {};
        c = __builtin_amdgcn_wmma_f32_16x16x32_f16(
                false, a, false, ones, (short)0, c, false, false);

        // D layout: c[r] = W_{M}, M = r (lanes 0-15) / 8+r (lanes 16-31),
        // identical across columns. Column-0 lanes park rows in LDS.
        if ((lane & 15) == 0) {
            const int base = r0 + ((lane < 16) ? 0 : 8);
            *reinterpret_cast<float4*>(&sW[base])     = make_float4(c[0], c[1], c[2], c[3]);
            *reinterpret_cast<float4*>(&sW[base + 4]) = make_float4(c[4], c[5], c[6], c[7]);
        }
        // Linear-recurrence coefficients, anchored at tile-end time t[r0+15]:
        //   H_p = decay_p * H_{p-1} + L_p,  L_p = W_15 + 1 (self term).
        if (lane == 16) {
            const float tend = t[r0 + 15];
            sAb[p] = c[7] + 1.0f;
            sAa[p] = (p > 0) ? __expf(-beta * (tend - t[r0 - 1])) : 0.0f;
        }
    }
    __syncthreads();

    // ---------------- Phase 2: Kogge-Stone linear-recurrence scan ----------------
    // (a2,b2) o (a1,b1) = (a2*a1, a2*b1 + b2); identity H_{-1} = 0.
    for (int s = 1; s < nblk; s <<= 1) {
        float ap = 0.0f, bp = 0.0f, amy = 0.0f, bmy = 0.0f;
        const bool has = (tid < nblk) && (tid >= s);
        if (has)        { ap  = sAa[tid - s]; bp  = sAb[tid - s]; }
        if (tid < nblk) { amy = sAa[tid];     bmy = sAb[tid];     }
        __syncthreads();
        if (has) { sAa[tid] = amy * ap; sAb[tid] = amy * bp + bmy; }
        __syncthreads();
    }
    // sAb[p] == H_p

    // ---------------- Phase 3: per-event log-likelihood ----------------
    float accL = 0.0f, accM = 0.0f;
    for (int e = tid; e < N; e += NTHREADS) {
        const int   p   = e >> 4;
        const float te  = t[e];
        float cross = 0.0f;
        if (p > 0) {
            const float tpe = t[(p << 4) - 1];    // previous tile's end time
            cross = __expf(-beta * (te - tpe)) * sAb[p - 1];
        }
        const float lamb = ab * (sW[e] + cross) + mu;
        const float mki  = mk[e];
        accL += mki * __logf(lamb + 1e-8f);
        accM += mki;
    }
    accL = wave_sum32(accL);
    accM = wave_sum32(accM);
    if (lane == 0) {
        atomicAdd(&sLog, accL);                   // ds_add_f32
        atomicAdd(&sMsk, accM);
    }
    __syncthreads();

    if (tid == 0) {
        const float t1v   = t1a[b];
        const float t0v   = t0a[b];
        const float Hfin  = sAb[nblk - 1];
        const float tlast = t[N - 1];
        // sum_j exp(-beta*(t1 - t_j)) = exp(-beta*(t1 - t_{N-1})) * H_final
        const float tail = __expf(-beta * (t1v - tlast)) * Hfin;
        const float comp = (t1v - t0v) * mu - alpha * (tail - sMsk);
        out[b] = sLog - comp;
    }
}

extern "C" void kernel_launch(void* const* d_in, const int* in_sizes, int n_in,
                              void* d_out, int out_size, void* d_ws, size_t ws_size,
                              hipStream_t stream) {
    const float* event_times = (const float*)d_in[0];  // [B, N]
    const float* input_mask  = (const float*)d_in[1];  // [B, N]
    const float* t0          = (const float*)d_in[2];  // [B]
    const float* t1          = (const float*)d_in[3];  // [B]
    const float* mu          = (const float*)d_in[4];  // [1]
    const float* alpha       = (const float*)d_in[5];  // [1]
    const float* beta        = (const float*)d_in[6];  // [1]
    float* out = (float*)d_out;                        // [B]

    const int B = in_sizes[2];
    const int N = in_sizes[0] / B;                     // 4096 (multiple of 16, <= MAXN)

    hawkes_loglik_kernel<<<dim3(B), dim3(NTHREADS), 0, stream>>>(
        event_times, input_mask, t0, t1, mu, alpha, beta, out, N);
}
